// LinearAttentionMem_89026082111940
// MI455X (gfx1250) — matrix operand order repaired and checked by reading, not compiled, not run on hardware
//
#include <hip/hip_runtime.h>

// ---------------------------------------------------------------------------
// LinearAttentionMem on MI455X (gfx1250, wave32, WMMA bf16 + TDM async copy)
//
//   out[b,h] = sigmoid(x@Wf^T+bf).mean(S) * M[b,h] + (K_h^T V_h)/S
//   K = x@Wk^T+bk, V = x@Wv^T+bv   (per-head 128-wide slices)
//
// Compute-bound (~550 GFLOP vs ~300 MiB HBM) -> bf16 WMMA, f32 accumulate.
// Fully fused; W tiles staged by Tensor Data Mover (double-buffered,
// s_wait_tensorcnt), X tiles by VALU (needs f32->bf16 convert).
// ---------------------------------------------------------------------------

#define B_      8
#define S_      4096
#define DM_     2048
#define H_      16
#define D_      128
#define SPLIT   4                 // S split for kv partials
#define SCHUNK  (S_ / SPLIT)      // 1024 rows per WG
#define SB      128               // s-rows per projection block
#define CB      64                // contraction (c) chunk
#define NC      (DM_ / CB)        // 32 chunks
#define XP      (CB + 8)          // padded LDS pitch (72 halfwords)
#define TP      (SB + 8)          // padded LDS pitch for transposed K/V
#define FBLK    (B_ * S_ / 256)   // 128 forget-gate blocks

typedef unsigned short u16;
typedef __attribute__((ext_vector_type(16))) __bf16        v16bf;
typedef __attribute__((ext_vector_type(8)))  float         v8f;
typedef __attribute__((ext_vector_type(4))) unsigned int   ui32x4;
typedef __attribute__((ext_vector_type(8)))  int           i32x8;
typedef __attribute__((ext_vector_type(4)))  int           i32x4;
typedef __attribute__((ext_vector_type(4)))  float         f32x4;

union Frag { v16bf v; ui32x4 q[2]; };

__device__ __forceinline__ u16 f2bf(float f) {  // round-to-nearest-even
  unsigned u = __float_as_uint(f);
  return (u16)((u + 0x7FFFu + ((u >> 16) & 1u)) >> 16);
}

__device__ __forceinline__ v8f zero8() {
  v8f z;
#pragma unroll
  for (int i = 0; i < 8; ++i) z[i] = 0.0f;
  return z;
}

// A-matrix fragment (16x32 bf16): lane m=lane&15; two 16B chunks
__device__ __forceinline__ v16bf ldsA(const u16* base, int pitch, int mrow,
                                      int kb, int lane) {
  const u16* p = base + (mrow + (lane & 15)) * pitch + kb + ((lane >> 4) << 3);
  Frag f;
  f.q[0] = *(const ui32x4*)(p);
  f.q[1] = *(const ui32x4*)(p + 16);
  return f.v;
}

// B-matrix fragment (32x16 bf16): lane n=lane&15; two 16B chunks (same K order as A)
__device__ __forceinline__ v16bf ldsB(const u16* base, int pitch, int ncol,
                                      int kb, int lane) {
  const u16* p = base + (ncol + (lane & 15)) * pitch + kb + ((lane >> 4) << 3);
  Frag f;
  f.q[0] = *(const ui32x4*)(p);
  f.q[1] = *(const ui32x4*)(p + 16);
  return f.v;
}

__device__ __forceinline__ v8f wmma_bf16(v16bf a, v16bf b, v8f c) {
  return __builtin_amdgcn_wmma_f32_16x16x32_bf16(false, a, false, b,
                                                 (short)0, c, false, false);
}

// ---------------------------------------------------------------------------
// TDM: DMA one 128-row x 64-col bf16 tile (row stride DM_) into LDS,
// padding 4 DWORDs every 32 DWORDs (-> pitch 72 halfwords = XP).
// D# group0: count=1, lds_addr, 57b global_addr, type=2.
// D# group1: data_size=2B, pad_enable, pad_interval=32DW(code 4),
//            pad_amount=4DW(code 3), tensor_dim0=2048, tensor_dim1=128,
//            tile_dim0=64, tile_dim1=128, tensor_dim0_stride=2048.
// Toolchain uses the 6-arg builtin: (g0, g1, g2, g3, g4, cpol).
// ---------------------------------------------------------------------------
__device__ __forceinline__ void tdm_load_w(const u16* gsrc, unsigned lds_off) {
  const unsigned long long ga = (unsigned long long)(uintptr_t)gsrc;
  ui32x4 g0;
  g0[0] = 1u;                                                  // count=1
  g0[1] = lds_off;                                             // lds_addr
  g0[2] = (unsigned)ga;                                        // gaddr[31:0]
  g0[3] = (unsigned)((ga >> 32) & 0x1FFFFFFull) | (2u << 30);  // gaddr hi|type
  i32x8 g1;
  g1[0] = (1 << 16) | (1 << 20) | (4 << 22) | (3 << 25);
  g1[1] = (int)(2048u << 16);   // tensor_dim0 = 2048
  g1[2] = (int)(128u << 16);    // tensor_dim1 = 128
  g1[3] = (int)(64u << 16);     // tile_dim0   = 64
  g1[4] = 128;                  // tile_dim1   = 128
  g1[5] = DM_;                  // tensor_dim0_stride = 2048
  g1[6] = 0;
  g1[7] = 0;
  i32x4 z4;
  z4[0] = 0; z4[1] = 0; z4[2] = 0; z4[3] = 0;
  i32x8 z8;
#pragma unroll
  for (int i = 0; i < 8; ++i) z8[i] = 0;
  __builtin_amdgcn_tensor_load_to_lds(g0, g1, z4, z4, z8, 0);
}

// ---------------------------------------------------------------------------
// Prep: convert Wk/Wv to bf16 once (halves the L2-resident weight traffic)
// ---------------------------------------------------------------------------
__global__ void prep_w_kernel(const float* __restrict__ Wk,
                              const float* __restrict__ Wv,
                              u16* __restrict__ wkbf, u16* __restrict__ wvbf) {
  const int n = DM_ * DM_;
  for (int i = blockIdx.x * blockDim.x + threadIdx.x; i < n;
       i += gridDim.x * blockDim.x) {
    wkbf[i] = f2bf(Wk[i]);
    wvbf[i] = f2bf(Wv[i]);
  }
}

// ---------------------------------------------------------------------------
// Forget gate: per block of 256 rows, sum_h sigmoid(x_row . Wf_h + bf_h)
// ---------------------------------------------------------------------------
__global__ void __launch_bounds__(256)
forget_kernel(const float* __restrict__ x, const float* __restrict__ Wf,
              const float* __restrict__ bfv, float* __restrict__ fpart) {
  __shared__ float red[H_];
  const int tid = threadIdx.x;
  if (tid < H_) red[tid] = 0.0f;
  __syncthreads();

  const int row = blockIdx.x * 256 + tid;  // all rows of a block share b
  const float* xr = x + (size_t)row * DM_;
  float acc[H_];
#pragma unroll
  for (int hh = 0; hh < H_; ++hh) acc[hh] = 0.0f;
  for (int c = 0; c < DM_; ++c) {
    const float xv = xr[c];
#pragma unroll
    for (int hh = 0; hh < H_; ++hh) acc[hh] += xv * Wf[hh * DM_ + c];
  }
#pragma unroll
  for (int hh = 0; hh < H_; ++hh) {
    const float s = 1.0f / (1.0f + __expf(-(acc[hh] + bfv[hh])));
    atomicAdd(&red[hh], s);
  }
  __syncthreads();
  if (tid < H_) fpart[blockIdx.x * H_ + tid] = red[tid];
}

// ---------------------------------------------------------------------------
// Fused projection + kv accumulation.
// Grid: SPLIT * H * B workgroups of 256 threads (8 waves).
// Waves 0-3: K projection (32x128 strip each); waves 4-7: V projection.
// All 8 waves: one 32x64 quadrant of the 128x128 kv accumulator.
// W tiles arrive by double-buffered TDM; wave 0 drives the DMA pipeline.
// ---------------------------------------------------------------------------
__global__ void __launch_bounds__(256, 1)
fused_kv_kernel(const float* __restrict__ x, const u16* __restrict__ wkbf,
                const u16* __restrict__ wvbf, const float* __restrict__ bk,
                const float* __restrict__ bv, float* __restrict__ kvpart) {
  __shared__ u16 Xs[SB * XP];          // x tile, bf16
  __shared__ u16 Wks[2][D_ * XP];      // Wk head-slice tile (double buffer)
  __shared__ u16 Wvs[2][D_ * XP];      // Wv head-slice tile (double buffer)
  __shared__ u16 Kts[D_ * TP];         // K^T (d-major) bf16
  __shared__ u16 Vts[D_ * TP];         // V^T (d-major) bf16

  const int tid  = threadIdx.x;
  const int lane = tid & 31;
  const int w    = tid >> 5;

  const int bx = blockIdx.x;
  const int sc = bx & (SPLIT - 1);
  const int h  = (bx >> 2) & (H_ - 1);
  const int b  = bx >> 6;
  const int hbase = h * D_;

  const bool   isV  = (w >= 4);
  const int    m0A  = (w & 3) * 32;         // projection output row strip
  const float* bias = isV ? bv : bk;
  u16*         Tdst = isV ? Vts : Kts;

  const int m0B = (w & 3) * 32;             // kv tile origin (dk)
  const int n0B = (w >> 2) * 64;            // kv tile origin (dv)

  const u16* wkh = wkbf + (size_t)hbase * DM_;   // head slice bases
  const u16* wvh = wvbf + (size_t)hbase * DM_;
  const unsigned wkOff[2] = {(unsigned)(uintptr_t)&Wks[0][0],
                             (unsigned)(uintptr_t)&Wks[1][0]};
  const unsigned wvOff[2] = {(unsigned)(uintptr_t)&Wvs[0][0],
                             (unsigned)(uintptr_t)&Wvs[1][0]};

  v8f accB[2][4];
#pragma unroll
  for (int i = 0; i < 2; ++i)
#pragma unroll
    for (int j = 0; j < 4; ++j) accB[i][j] = zero8();

  for (int sb = 0; sb < SCHUNK / SB; ++sb) {
    const int srow0 = sc * SCHUNK + sb * SB;
    v8f accA[2][8];
#pragma unroll
    for (int i = 0; i < 2; ++i)
#pragma unroll
      for (int j = 0; j < 8; ++j) accA[i][j] = zero8();

    if (w == 0) {                       // DMA prologue: chunk 0 -> buffer 0
      tdm_load_w(wkh, wkOff[0]);
      tdm_load_w(wvh, wvOff[0]);
    }

    for (int cc = 0; cc < NC; ++cc) {
      const int c0  = cc * CB;
      const int cur = cc & 1;
      __syncthreads();  // prior compute done: W bufs + Xs reusable
      if (w == 0) {
        if (cc + 1 < NC) {              // prefetch next chunk into other buf
          tdm_load_w(wkh + c0 + CB, wkOff[1 - cur]);
          tdm_load_w(wvh + c0 + CB, wvOff[1 - cur]);
          __builtin_amdgcn_s_wait_tensorcnt((short)2);  // current pair done
        } else {
          __builtin_amdgcn_s_wait_tensorcnt((short)0);
        }
      }
      // ---- stage X tile (128x64 f32 -> bf16), 8 float4 per thread ----
#pragma unroll
      for (int i = 0; i < 8; ++i) {
        const int idx = tid + i * 256;            // 0..2047
        const int r = idx >> 4, q = idx & 15;
        const float* gp =
            x + (size_t)(b * S_ + srow0 + r) * DM_ + c0 + q * 4;
        if (i < 2 && cc + 1 < NC) __builtin_prefetch(gp + CB, 0, 1);
        const f32x4 xv = *(const f32x4*)gp;
        u16* d = &Xs[r * XP + q * 4];
        d[0] = f2bf(xv.x); d[1] = f2bf(xv.y);
        d[2] = f2bf(xv.z); d[3] = f2bf(xv.w);
      }
      __syncthreads();  // X staged + TDM data visible to all waves
      // ---- projection WMMA, B-fragments software-pipelined ----
      const u16* wl = isV ? Wvs[cur] : Wks[cur];
#pragma unroll
      for (int kk = 0; kk < 2; ++kk) {
        const int kb = kk * 32;
        const v16bf a0 = ldsA(Xs, XP, m0A, kb, lane);
        const v16bf a1 = ldsA(Xs, XP, m0A + 16, kb, lane);
        v16bf bc = ldsB(wl, XP, 0, kb, lane);
#pragma unroll
        for (int nt = 0; nt < 8; ++nt) {
          const v16bf bn =
              (nt < 7) ? ldsB(wl, XP, (nt + 1) * 16, kb, lane) : bc;
          accA[0][nt] = wmma_bf16(a0, bc, accA[0][nt]);
          accA[1][nt] = wmma_bf16(a1, bc, accA[1][nt]);
          bc = bn;
        }
      }
    }
    __syncthreads();
    // ---- bias + transposed bf16 store: Tdst[d][s] (padded pitch) ----
#pragma unroll
    for (int rt = 0; rt < 2; ++rt) {
#pragma unroll
      for (int nt = 0; nt < 8; ++nt) {
        const int n   = nt * 16 + (lane & 15);
        const float bsv = bias[hbase + n];
#pragma unroll
        for (int r = 0; r < 8; ++r) {
          const int srow = m0A + rt * 16 + r + ((lane >> 4) << 3);
          Tdst[n * TP + srow] = f2bf(accA[rt][nt][r] + bsv);
        }
      }
    }
    __syncthreads();
    // ---- kv += K^T V : A = Kts[dk][s], B = Vts[dv][s], K-dim = 128 ----
#pragma unroll
    for (int kk = 0; kk < 4; ++kk) {
      const int kb = kk * 32;
      const v16bf a0 = ldsA(Kts, TP, m0B, kb, lane);
      const v16bf a1 = ldsA(Kts, TP, m0B + 16, kb, lane);
      v16bf bc = ldsB(Vts, TP, n0B, kb, lane);
#pragma unroll
      for (int nt = 0; nt < 4; ++nt) {
        const v16bf bn =
            (nt < 3) ? ldsB(Vts, TP, n0B + (nt + 1) * 16, kb, lane) : bc;
        accB[0][nt] = wmma_bf16(a0, bc, accB[0][nt]);
        accB[1][nt] = wmma_bf16(a1, bc, accB[1][nt]);
        bc = bn;
      }
    }
  }
  // ---- write f32 partials: kvpart[sc][b][h][dk][dv] ----
  float* dst = kvpart + (size_t)((sc * B_ + b) * H_ + h) * D_ * D_;
#pragma unroll
  for (int rt = 0; rt < 2; ++rt)
#pragma unroll
    for (int nt = 0; nt < 4; ++nt)
#pragma unroll
      for (int r = 0; r < 8; ++r) {
        const int dk = m0B + rt * 16 + r + ((lane >> 4) << 3);
        const int dv = n0B + nt * 16 + (lane & 15);
        dst[dk * D_ + dv] = accB[rt][nt][r];
      }
}

// ---------------------------------------------------------------------------
// Combine: out = forget * M + (sum of partials)/S
// ---------------------------------------------------------------------------
__global__ void __launch_bounds__(256)
combine_kernel(const float* __restrict__ kvpart, const float* __restrict__ fpart,
               const float* __restrict__ M, float* __restrict__ out) {
  const int bh = blockIdx.x;               // b*H + h
  const int b = bh >> 4, h = bh & 15;
  float fsum = 0.0f;
  for (int j = 0; j < S_ / 256; ++j)       // 16 forget blocks per batch
    fsum += fpart[(b * (S_ / 256) + j) * H_ + h];
  const float forget = fsum / (float)S_;
  const float inv_s  = 1.0f / (float)S_;
  const size_t base = (size_t)bh * D_ * D_;
  for (int i = threadIdx.x; i < D_ * D_; i += 256) {
    float s = 0.0f;
#pragma unroll
    for (int scc = 0; scc < SPLIT; ++scc)
      s += kvpart[((size_t)(scc * B_ + b) * H_ + h) * D_ * D_ + i];
    out[base + i] = forget * M[base + i] + s * inv_s;
  }
}

// ---------------------------------------------------------------------------
extern "C" void kernel_launch(void* const* d_in, const int* in_sizes, int n_in,
                              void* d_out, int out_size, void* d_ws,
                              size_t ws_size, hipStream_t stream) {
  const float* x   = (const float*)d_in[0];
  const float* Wk  = (const float*)d_in[1];
  const float* bk  = (const float*)d_in[2];
  const float* Wv  = (const float*)d_in[3];
  const float* bv  = (const float*)d_in[4];
  const float* Wf  = (const float*)d_in[5];
  const float* bfv = (const float*)d_in[6];
  const float* M   = (const float*)d_in[7];
  float* out = (float*)d_out;

  char* ws = (char*)d_ws;
  u16*   wkbf   = (u16*)(ws);                                  //  8 MiB
  u16*   wvbf   = (u16*)(ws + (size_t)8 * 1024 * 1024);        //  8 MiB
  float* kvpart = (float*)(ws + (size_t)16 * 1024 * 1024);     // 32 MiB
  float* fpart  = (float*)(ws + (size_t)48 * 1024 * 1024);     // 8 KiB

  prep_w_kernel<<<2048, 256, 0, stream>>>(Wk, Wv, wkbf, wvbf);
  forget_kernel<<<FBLK, 256, 0, stream>>>(x, Wf, bfv, fpart);
  fused_kv_kernel<<<SPLIT * H_ * B_, 256, 0, stream>>>(x, wkbf, wvbf, bk, bv,
                                                       kvpart);
  combine_kernel<<<B_ * H_, 256, 0, stream>>>(kvpart, fpart, M, out);
}
